// TemporalGNN_69612829933740
// MI455X (gfx1250) — matrix operand order repaired and checked by reading, not compiled, not run on hardware
//
#include <hip/hip_runtime.h>
#include <hip/hip_bf16.h>

// ---------------- problem constants ----------------
#define BB   8
#define TT   16
#define NN   2048
#define EE   32768
#define E2   (EE + NN)        // 34816 (edges + self loops)
#define FIN  32
#define CC   32
#define BT   (BB * TT)        // 128
#define ROWS (BB * TT * NN)   // 262144 (b,t,n) rows

typedef __attribute__((ext_vector_type(16))) _Float16 v16h;
typedef __attribute__((ext_vector_type(8)))  float    v8f;

static __device__ __forceinline__ v8f wmma_f16(v16h a, v16h b, v8f c) {
  // D = A(16x32 f16) * B(32x16 f16) + C(16x16 f32)
  return __builtin_amdgcn_wmma_f32_16x16x32_f16(
      false, a, false, b, (short)0, c, false, false);
}

// order-preserving float <-> uint for atomicMax-based segment max
static __device__ __forceinline__ unsigned f2ord(float f) {
  unsigned u = __float_as_uint(f);
  return (u & 0x80000000u) ? ~u : (u | 0x80000000u);
}
static __device__ __forceinline__ float ord2f(unsigned u) {
  u = (u & 0x80000000u) ? (u & 0x7fffffffu) : ~u;
  return __uint_as_float(u);
}

static __device__ __forceinline__ void edge_nodes(const int* ei, int e, int& s, int& d) {
  if (e < EE) { s = ei[e]; d = ei[EE + e]; }
  else        { s = e - EE; d = e - EE; }      // appended self loop
}

// ---------------- init kernels ----------------
__global__ void k_init_nodes(unsigned* smax, float* denom, float* deg, float* attr_acc) {
  int i = blockIdx.x * blockDim.x + threadIdx.x;
  if (i < NN * BT) { smax[i] = 0x007FFFFFu /* f2ord(-inf) */; denom[i] = 0.0f; }
  if (i < NN) { deg[i] = 0.0f; attr_acc[i] = 0.0f; }
}

__global__ void k_init_gat(float* gat, const float* out_bias) {
  int i = blockIdx.x * blockDim.x + threadIdx.x;
  if (i < ROWS * CC) gat[i] = out_bias[i & 31];
}

__global__ void k_edge_deg(const int* ei, const float* edge_attr,
                           float* deg, float* attr_a) {
  int e = blockIdx.x * blockDim.x + threadIdx.x;
  if (e >= EE) return;
  int d = ei[EE + e];
  atomicAdd(&deg[d], 1.0f);
  atomicAdd(&attr_a[EE + d], edge_attr[e]);  // accumulate into self-loop slot
  attr_a[e] = edge_attr[e];
}

__global__ void k_mean_attr(float* attr_a, const float* deg) {
  int n = blockIdx.x * blockDim.x + threadIdx.x;
  if (n >= NN) return;
  attr_a[EE + n] = attr_a[EE + n] / fmaxf(deg[n], 1.0f);
}

// ---------------- weight packing into WMMA B-operand layout ----------------
// B layout (16-bit, 32x16): lane l -> column n = (l&15), K-half h = (l>>4);
// halves j=0..7 hold B[h*8 + j][n], j=8..15 hold B[16 + h*8 + j-8][n]. 16 tiles of 512 halves:
//   tiles 0-1 : Wl  (B[k][n] = Wl[k*CC+n])
//   tiles 2-3 : Wr
//   tiles 4-9 : w_ih^T (B[k][n] = w_ih[n*CC+k], n in 0..95)
//   tiles 10-15: w_hh^T
__global__ void k_pack_weights(const float* Wl, const float* Wr,
                               const float* w_ih, const float* w_hh,
                               _Float16* packW) {
  int tid = blockIdx.x * blockDim.x + threadIdx.x;
  if (tid >= 16 * 512) return;
  int tile = tid >> 9, rem = tid & 511;
  int lane = rem >> 4, j = rem & 15;
  int h = lane >> 4, nloc = lane & 15;
  int k = (j < 8) ? (h * 8 + j) : (16 + h * 8 + (j - 8));
  float v;
  if (tile < 2)       { int n = tile * 16 + nloc;        v = Wl[k * CC + n]; }
  else if (tile < 4)  { int n = (tile - 2) * 16 + nloc;  v = Wr[k * CC + n]; }
  else if (tile < 10) { int n = (tile - 4) * 16 + nloc;  v = w_ih[n * CC + k]; }
  else                { int n = (tile - 10) * 16 + nloc; v = w_hh[n * CC + k]; }
  packW[tid] = (_Float16)v;
}

static __device__ __forceinline__ v16h load_btile(const _Float16* packW, int tile, int lane) {
  return *(const v16h*)(packW + tile * 512 + lane * 16);
}

// A layout (16-bit, 16x32): lane l -> row m=(l&15), h=(l>>4);
// halves 0..7 = A[m][h*8 + 0..7], halves 8..15 = A[m][16 + h*8 + 0..7].
static __device__ __forceinline__ v16h load_a_f32(const float* row, int h) {
  v16h a;
#pragma unroll
  for (int i = 0; i < 8; ++i) {
    a[i]     = (_Float16)row[h * 8 + i];
    a[8 + i] = (_Float16)row[16 + h * 8 + i];
  }
  return a;
}

// ---------------- xl / xr projection (WMMA) ----------------
// wave per 16-row tile; K = FIN = 32 -> one WMMA per 16x16 output tile.
__global__ void k_xlxr(const float* __restrict__ x, const _Float16* __restrict__ packW,
                       const float* __restrict__ bl, const float* __restrict__ br,
                       float* __restrict__ xl, float* __restrict__ xr) {
  int lane = threadIdx.x & 31, wave = threadIdx.x >> 5;
  int tile = blockIdx.x * 8 + wave;
  if (tile >= ROWS / 16) return;
  int row0 = tile * 16;
  int h = lane >> 4, nloc = lane & 15;

  // A-operand row for this lane is m = lane & 15 (== nloc)
  v16h a = load_a_f32(x + (size_t)(row0 + nloc) * FIN, h);
  v16h bWl0 = load_btile(packW, 0, lane), bWl1 = load_btile(packW, 1, lane);
  v16h bWr0 = load_btile(packW, 2, lane), bWr1 = load_btile(packW, 3, lane);
  v8f z = {};
  v8f cl0 = wmma_f16(a, bWl0, z), cl1 = wmma_f16(a, bWl1, z);
  v8f cr0 = wmma_f16(a, bWr0, z), cr1 = wmma_f16(a, bWr1, z);

  float bl0 = bl[nloc], bl1 = bl[16 + nloc], br0 = br[nloc], br1 = br[16 + nloc];
#pragma unroll
  for (int r = 0; r < 8; ++r) {
    int mr = h * 8 + r;                         // D layout: lanes16-31 hold M=r+8
    size_t ro = (size_t)(row0 + mr) * CC;
    xl[ro + nloc]      = cl0[r] + bl0;
    xl[ro + nloc + 16] = cl1[r] + bl1;
    xr[ro + nloc]      = cr0[r] + br0;
    xr[ro + nloc + 16] = cr1[r] + br1;
  }
}

// ---------------- edge score: s = att . leaky_relu(xl[src]+xr[dst]+eemb) ----------------
__global__ void k_edge_s(const int* __restrict__ ei, const float* __restrict__ attr_a,
                         const float* __restrict__ We, const float* __restrict__ att,
                         const float* __restrict__ xl, const float* __restrict__ xr,
                         float* __restrict__ sbuf, unsigned* __restrict__ smax) {
  int lane = threadIdx.x & 31;
  long long gid = (long long)blockIdx.x * 8 + (threadIdx.x >> 5);
  if (gid >= (long long)E2 * BT) return;
  int bt = (int)(gid / E2), e = (int)(gid % E2);
  int src, dst; edge_nodes(ei, e, src, dst);
  float v = xl[((size_t)bt * NN + src) * CC + lane]
          + xr[((size_t)bt * NN + dst) * CC + lane]
          + attr_a[e] * We[lane];
  v = v > 0.0f ? v : 0.2f * v;                  // leaky_relu(0.2)
  float p = v * att[lane];
#pragma unroll
  for (int o = 16; o; o >>= 1) p += __shfl_xor(p, o, 32);
  if (lane == 0) {
    sbuf[(size_t)bt * E2 + e] = p;
    atomicMax(&smax[bt * NN + dst], f2ord(p));
  }
}

// ---------------- softmax numerator + denominator ----------------
__global__ void k_edge_exp(const int* __restrict__ ei, const unsigned* __restrict__ smax,
                           float* __restrict__ sbuf, float* __restrict__ denom) {
  long long idx = (long long)blockIdx.x * blockDim.x + threadIdx.x;
  if (idx >= (long long)E2 * BT) return;
  int bt = (int)(idx / E2), e = (int)(idx % E2);
  int src, dst; edge_nodes(ei, e, src, dst); (void)src;
  float sm = ord2f(smax[bt * NN + dst]);
  float es = __expf(sbuf[(size_t)bt * E2 + e] - sm);
  sbuf[(size_t)bt * E2 + e] = es;
  atomicAdd(&denom[bt * NN + dst], es);
}

// ---------------- alpha-weighted scatter: gat[dst] += alpha * xl[src] ----------------
__global__ void k_edge_scatter(const int* __restrict__ ei, const float* __restrict__ sbuf,
                               const float* __restrict__ denom, const float* __restrict__ xl,
                               float* __restrict__ gat) {
  int lane = threadIdx.x & 31;
  long long gid = (long long)blockIdx.x * 8 + (threadIdx.x >> 5);
  if (gid >= (long long)E2 * BT) return;
  int bt = (int)(gid / E2), e = (int)(gid % E2);
  int src, dst; edge_nodes(ei, e, src, dst);
  float alpha = sbuf[(size_t)bt * E2 + e] / denom[bt * NN + dst];
  float val = xl[((size_t)bt * NN + src) * CC + lane] * alpha;
  atomicAdd(&gat[((size_t)bt * NN + dst) * CC + lane], val);
}

// ---------------- GRU (WMMA, 12 tiles per timestep) ----------------
// wave per 16 (b,n) rows; h kept in D-layout registers; LDS transpose -> A layout.
// Next timestep's 2KB input tile is double-buffered into LDS with
// global_load_async_to_lds_b128 (ASYNCcnt), overlapping the fetch with the
// 12 WMMAs + gate transcendentals of the current step.
__global__ void k_gru(const float* __restrict__ gat, const _Float16* __restrict__ packW,
                      const float* __restrict__ b_ih, const float* __restrict__ b_hh,
                      float* __restrict__ y) {
  __shared__ float xsh[8][2][512];               // per-wave double buffer: 16x32 f32 tile
  __shared__ float hsh[8][16][33];
  int lane = threadIdx.x & 31, wave = threadIdx.x >> 5;
  int tile = blockIdx.x * 8 + wave;              // 1024 tiles exactly
  int b = tile >> 7, n0 = (tile & 127) * 16;
  int h = lane >> 4, nloc = lane & 15;

  // async copy of the (contiguous) 16x32 f32 input tile for timestep t into buf
  auto issue_copy = [&](int t, int buf) {
    const char* g = (const char*)(gat + ((size_t)((b * TT + t) * NN) + n0) * CC);
    // low 32 bits of a generic pointer into LDS == wave-relative LDS address
    unsigned lbase = (unsigned)(size_t)(const void*)&xsh[wave][buf][0];
#pragma unroll
    for (int j = 0; j < 4; ++j) {
      unsigned loff = lbase + (unsigned)(j * 512 + lane * 16);
      unsigned long long ga = (unsigned long long)(size_t)(g + j * 512 + lane * 16);
      asm volatile("global_load_async_to_lds_b128 %0, %1, off"
                   :: "v"(loff), "v"(ga) : "memory");
    }
  };

  // persistent B operands: w_ih tiles 4..9, w_hh tiles 10..15
  v16h Bi0 = load_btile(packW, 4, lane),  Bi1 = load_btile(packW, 5, lane);
  v16h Bi2 = load_btile(packW, 6, lane),  Bi3 = load_btile(packW, 7, lane);
  v16h Bi4 = load_btile(packW, 8, lane),  Bi5 = load_btile(packW, 9, lane);
  v16h Bh0 = load_btile(packW, 10, lane), Bh1 = load_btile(packW, 11, lane);
  v16h Bh2 = load_btile(packW, 12, lane), Bh3 = load_btile(packW, 13, lane);
  v16h Bh4 = load_btile(packW, 14, lane), Bh5 = load_btile(packW, 15, lane);

  float bi0 = b_ih[nloc],      bi1 = b_ih[16 + nloc];
  float bi2 = b_ih[32 + nloc], bi3 = b_ih[48 + nloc];
  float bi4 = b_ih[64 + nloc], bi5 = b_ih[80 + nloc];
  float bh0 = b_hh[nloc],      bh1 = b_hh[16 + nloc];
  float bh2 = b_hh[32 + nloc], bh3 = b_hh[48 + nloc];
  float bh4 = b_hh[64 + nloc], bh5 = b_hh[80 + nloc];

  v8f hlo = {}, hhi = {};                        // h state, D layout (lo cols 0-15, hi 16-31)

  issue_copy(0, 0);                              // prologue: fetch t=0 tile

  for (int t = 0; t < TT; ++t) {
    int buf = t & 1;
    // wait for this step's async tile (and any prior) to land in LDS
    asm volatile("s_wait_asynccnt 0x0" ::: "memory");

    // --- A operand from GRU input x_t (LDS tile, row-major 16x32 f32) ---
    const float* xs = &xsh[wave][buf][0];
    v16h ax;
#pragma unroll
    for (int i = 0; i < 8; ++i) {
      ax[i]     = (_Float16)xs[nloc * 32 + h * 8 + i];
      ax[8 + i] = (_Float16)xs[nloc * 32 + 16 + h * 8 + i];
    }

    // kick off the next timestep's copy; it overlaps the WMMAs below
    if (t + 1 < TT) issue_copy(t + 1, buf ^ 1);

    // --- A operand from h state via LDS transpose ---
#pragma unroll
    for (int r = 0; r < 8; ++r) {
      hsh[wave][h * 8 + r][nloc]      = hlo[r];
      hsh[wave][h * 8 + r][nloc + 16] = hhi[r];
    }
    __syncthreads();
    v16h ah;
#pragma unroll
    for (int i = 0; i < 8; ++i) {
      ah[i]     = (_Float16)hsh[wave][nloc][h * 8 + i];
      ah[8 + i] = (_Float16)hsh[wave][nloc][16 + h * 8 + i];
    }
    __syncthreads();

    v8f z = {};
    v8f gi0 = wmma_f16(ax, Bi0, z), gi1 = wmma_f16(ax, Bi1, z);
    v8f gi2 = wmma_f16(ax, Bi2, z), gi3 = wmma_f16(ax, Bi3, z);
    v8f gi4 = wmma_f16(ax, Bi4, z), gi5 = wmma_f16(ax, Bi5, z);
    v8f gh0 = wmma_f16(ah, Bh0, z), gh1 = wmma_f16(ah, Bh1, z);
    v8f gh2 = wmma_f16(ah, Bh2, z), gh3 = wmma_f16(ah, Bh3, z);
    v8f gh4 = wmma_f16(ah, Bh4, z), gh5 = wmma_f16(ah, Bh5, z);

#pragma unroll
    for (int r = 0; r < 8; ++r) {
      // cols 0..15 (tiles 0/2/4)
      float rg = 1.0f / (1.0f + __expf(-((gi0[r] + bi0) + (gh0[r] + bh0))));
      float zg = 1.0f / (1.0f + __expf(-((gi2[r] + bi2) + (gh2[r] + bh2))));
      float nc = tanhf((gi4[r] + bi4) + rg * (gh4[r] + bh4));
      hlo[r] = (1.0f - zg) * nc + zg * hlo[r];
      // cols 16..31 (tiles 1/3/5)
      float rg2 = 1.0f / (1.0f + __expf(-((gi1[r] + bi1) + (gh1[r] + bh1))));
      float zg2 = 1.0f / (1.0f + __expf(-((gi3[r] + bi3) + (gh3[r] + bh3))));
      float nc2 = tanhf((gi5[r] + bi5) + rg2 * (gh5[r] + bh5));
      hhi[r] = (1.0f - zg2) * nc2 + zg2 * hhi[r];

      size_t ro = ((size_t)((b * TT + t) * NN) + n0 + h * 8 + r) * CC;
      y[ro + nloc]      = hlo[r];
      y[ro + nloc + 16] = hhi[r];
    }
  }
}

// ---------------- launcher ----------------
extern "C" void kernel_launch(void* const* d_in, const int* in_sizes, int n_in,
                              void* d_out, int out_size, void* d_ws, size_t ws_size,
                              hipStream_t stream) {
  const float* x         = (const float*)d_in[0];
  const int*   edge_index= (const int*)  d_in[1];
  const float* edge_attr = (const float*)d_in[2];
  const float* Wl        = (const float*)d_in[3];
  const float* bl        = (const float*)d_in[4];
  const float* Wr        = (const float*)d_in[5];
  const float* br        = (const float*)d_in[6];
  const float* We        = (const float*)d_in[7];
  const float* att       = (const float*)d_in[8];
  const float* out_bias  = (const float*)d_in[9];
  const float* w_ih      = (const float*)d_in[10];
  const float* w_hh      = (const float*)d_in[11];
  const float* b_ih      = (const float*)d_in[12];
  const float* b_hh      = (const float*)d_in[13];
  float* y = (float*)d_out;
  (void)in_sizes; (void)n_in; (void)out_size; (void)ws_size;

  char* ws = (char*)d_ws;
  size_t off = 0;
  auto carve = [&](size_t bytes) -> char* {
    char* p = ws + off; off += (bytes + 255) & ~(size_t)255; return p;
  };
  float*     xl     = (float*)    carve((size_t)ROWS * CC * 4);
  float*     xr     = (float*)    carve((size_t)ROWS * CC * 4);
  float*     gat    = (float*)    carve((size_t)ROWS * CC * 4);
  float*     sbuf   = (float*)    carve((size_t)E2 * BT * 4);
  unsigned*  smax   = (unsigned*) carve((size_t)NN * BT * 4);
  float*     denom  = (float*)    carve((size_t)NN * BT * 4);
  float*     attr_a = (float*)    carve((size_t)E2 * 4);
  float*     deg    = (float*)    carve((size_t)NN * 4);
  _Float16*  packW  = (_Float16*) carve((size_t)16 * 512 * 2);

  const int TB = 256;
  k_init_nodes<<<(NN * BT + TB - 1) / TB, TB, 0, stream>>>(smax, denom, deg, attr_a + EE);
  k_init_gat  <<<((size_t)ROWS * CC + TB - 1) / TB, TB, 0, stream>>>(gat, out_bias);
  k_edge_deg  <<<(EE + TB - 1) / TB, TB, 0, stream>>>(edge_index, edge_attr, deg, attr_a);
  k_mean_attr <<<(NN + TB - 1) / TB, TB, 0, stream>>>(attr_a, deg);
  k_pack_weights<<<(16 * 512 + TB - 1) / TB, TB, 0, stream>>>(Wl, Wr, w_ih, w_hh, packW);

  k_xlxr<<<(ROWS / 16 + 7) / 8, TB, 0, stream>>>(x, packW, bl, br, xl, xr);

  long long ebt = (long long)E2 * BT;              // 4,456,448
  k_edge_s      <<<(unsigned)(ebt / 8), TB, 0, stream>>>(edge_index, attr_a, We, att,
                                                         xl, xr, sbuf, smax);
  k_edge_exp    <<<(unsigned)(ebt / TB), TB, 0, stream>>>(edge_index, smax, sbuf, denom);
  k_edge_scatter<<<(unsigned)(ebt / 8), TB, 0, stream>>>(edge_index, sbuf, denom, xl, gat);

  k_gru<<<(BB * NN / 16) / 8, TB, 0, stream>>>(gat, packW, b_ih, b_hh, y);
}
